// SparseLayer_56349970923843
// MI455X (gfx1250) — matrix operand order repaired and checked by reading, not compiled, not run on hardware
//
#include <hip/hip_runtime.h>

// ---------------------------------------------------------------------------
// LCA sparse coding on MI455X (gfx1250).
//   B=512 images, D=4096 pixels, K=2048 filters.
//   excite = X @ F            [512 x 2048]   (bf16 WMMA, f32 accum)
//   inhibit = F^T @ F         [2048 x 2048]  (symmetric -> rows == columns)
//   50x { act = relu(u-0.25); u <- AdamW(u, -(du)) } with act@inhibit GEMM
// Everything is L2-resident (~50MB << 192MB), so V_WMMA_F32_16X16X32_BF16
// throughput is the ceiling. LDS tiles staged by the Tensor Data Mover.
// ---------------------------------------------------------------------------

#define SHRINK 0.25f
#define LRC    0.1f
#define WDC    0.01f
#define BETA1  0.9f
#define BETA2  0.999f
#define EPSC   1e-8f
#define NITER  50

#define BM 128
#define BN 128
#define BK 32
#define TILE_BYTES 8192            // 128 rows * 32 bf16 * 2B
#define BUF_BYTES  (2 * TILE_BYTES)

typedef __attribute__((ext_vector_type(16))) __bf16       v16bf;
typedef __attribute__((ext_vector_type(8)))  float        v8f;
typedef __attribute__((ext_vector_type(4)))  unsigned int u32x4v;
typedef __attribute__((ext_vector_type(8)))  int          i32x8v;
typedef __attribute__((ext_vector_type(4)))  int          i32x4v;

union FragBf { v16bf v; u32x4v q[2]; };

#if __has_builtin(__builtin_amdgcn_tensor_load_to_lds)
#define USE_TDM 1
#else
#define USE_TDM 0
#endif

#if USE_TDM
// 2D tile load via TDM. D# packing per CDNA5 ISA sec 8.3/8.4.
// data_size = 4B dwords; tile is (tile_h rows) x (16 dwords = 32 bf16).
// This toolchain exposes the 6-arg builtin:
//   (v4u32 g0, v8i32 g1, v4i32 g2, v4i32 g3, v8i32 extra, i32 cpol)
__device__ __forceinline__ void tdm_load_tile(unsigned lds_off,
                                              const __bf16* gsrc,
                                              unsigned tile_h,
                                              unsigned width_dw,   // row length & stride, dwords
                                              unsigned tensor_rows) {
  unsigned long long ga = (unsigned long long)(const void*)gsrc;
  u32x4v g0;
  g0[0] = 1u;                                          // count=1, user mode
  g0[1] = lds_off;                                     // LDS byte address
  g0[2] = (unsigned)ga;                                // global_addr[31:0]
  g0[3] = ((unsigned)(ga >> 32) & 0x01FFFFFFu) | (2u << 30); // addr[56:32]|type=2
  i32x8v g1;
  g1[0] = (int)(2u << 16);                             // wg_mask=0, data_size=2 (4B)
  g1[1] = (int)((width_dw & 0xFFFFu) << 16);           // tensor_dim0[15:0] @63:48
  g1[2] = (int)((width_dw >> 16) | ((tensor_rows & 0xFFFFu) << 16)); // dim0 hi | dim1 lo
  g1[3] = (int)((tensor_rows >> 16) | (16u << 16));    // dim1 hi | tile_dim0=16 dw
  g1[4] = (int)(tile_h & 0xFFFFu);                     // tile_dim1 rows, tile_dim2=0
  g1[5] = (int)width_dw;                               // tensor_dim0_stride[31:0]
  g1[6] = 0;                                           // stride hi, dim1_stride lo
  g1[7] = 0;
  i32x4v z4; z4[0] = z4[1] = z4[2] = z4[3] = 0;        // groups 2/3 unused (2D)
  i32x8v z8;
#pragma unroll
  for (int i = 0; i < 8; ++i) z8[i] = 0;
  __builtin_amdgcn_tensor_load_to_lds(g0, g1, z4, z4, z8, 0);
}
#endif

// C[m,n] = sum_k A[m,k] * Bt[n,k]   (A: Arows x Kdim bf16, Bt: Brows x Kdim bf16)
// MODE 0: store C as bf16 (inhibit)     MODE 1: store C as f32 (excite)
// MODE 2: fused LCA + AdamW epilogue, writes u/m/v (f32) and next act (bf16).
template <int MODE>
__global__ __launch_bounds__(256) void wmma_gemm(
    const __bf16* __restrict__ A, const __bf16* __restrict__ Bt,
    int Kdim, int Ncols, int Arows, int Brows,
    __bf16* __restrict__ outBf, float* __restrict__ outF,
    float* __restrict__ u, float* __restrict__ m, float* __restrict__ v,
    const float* __restrict__ excite, __bf16* __restrict__ actOut,
    float c1, float c2) {
  __shared__ __align__(16) char smem[2 * BUF_BYTES];

  const int tid  = threadIdx.x;
  const int lane = tid & 31;
  const int wid  = tid >> 5;
  const int wm   = wid & 3;          // 4 waves along M (32 rows each)
  const int wn   = wid >> 2;         // 2 waves along N (64 cols each)
  const int half = lane >> 4;
  const int lr   = lane & 15;

  const int m0 = blockIdx.y * BM;
  const int n0 = blockIdx.x * BN;
  const int KT = Kdim / BK;

  v8f acc[2][4];
  const v8f vzero = {0.f, 0.f, 0.f, 0.f, 0.f, 0.f, 0.f, 0.f};
#pragma unroll
  for (int i = 0; i < 2; ++i)
#pragma unroll
    for (int j = 0; j < 4; ++j) acc[i][j] = vzero;

#if USE_TDM
  const unsigned ldsBase = (unsigned)(size_t)(void*)smem;
  // --- pipelined: TDM stages tile kt+1 while waves compute on tile kt ---
  if (wid == 0) {
    tdm_load_tile(ldsBase, A + (size_t)m0 * Kdim, BM, (unsigned)(Kdim >> 1),
                  (unsigned)Arows);
    tdm_load_tile(ldsBase + TILE_BYTES, Bt + (size_t)n0 * Kdim, BN,
                  (unsigned)(Kdim >> 1), (unsigned)Brows);
  }
  for (int kt = 0; kt < KT; ++kt) {
    if (wid == 0) {
      if (kt + 1 < KT) {
        unsigned loff = ldsBase + (unsigned)(((kt + 1) & 1) * BUF_BYTES);
        tdm_load_tile(loff, A + (size_t)m0 * Kdim + (kt + 1) * BK, BM,
                      (unsigned)(Kdim >> 1), (unsigned)Arows);
        tdm_load_tile(loff + TILE_BYTES, Bt + (size_t)n0 * Kdim + (kt + 1) * BK,
                      BN, (unsigned)(Kdim >> 1), (unsigned)Brows);
        __builtin_amdgcn_s_wait_tensorcnt(2);  // current tile's 2 ops done
      } else {
        __builtin_amdgcn_s_wait_tensorcnt(0);
      }
    }
    __syncthreads();  // LDS buf[kt&1] ready for all waves
    {
      const char* As = smem + (kt & 1) * BUF_BYTES;
      const char* Bs = As + TILE_BYTES;
      FragBf a[2], b[4];
#pragma unroll
      for (int i = 0; i < 2; ++i) {
        const char* p = As + (wm * 32 + i * 16 + lr) * 64 + half * 16;
        a[i].q[0] = *(const u32x4v*)p;          // K = 8*half .. +8
        a[i].q[1] = *(const u32x4v*)(p + 32);   // K = 16+8*half .. +8
      }
#pragma unroll
      for (int j = 0; j < 4; ++j) {
        const char* p = Bs + (wn * 64 + j * 16 + lr) * 64 + half * 16;
        b[j].q[0] = *(const u32x4v*)p;          // K = 8*half .. +8
        b[j].q[1] = *(const u32x4v*)(p + 32);   // K = 16+8*half .. +8
      }
#pragma unroll
      for (int i = 0; i < 2; ++i)
#pragma unroll
        for (int j = 0; j < 4; ++j)
          acc[i][j] = __builtin_amdgcn_wmma_f32_16x16x32_bf16(
              false, a[i].v, false, b[j].v, (short)0, acc[i][j], false, false);
    }
    __syncthreads();  // all reads of buf[kt&1] done before it is re-staged
  }
#else
  // --- fallback: cooperative synchronous copy, no pipelining ---
  for (int kt = 0; kt < KT; ++kt) {
    char* As = smem;
    char* Bs = smem + TILE_BYTES;
    {
      int row = tid >> 1, seg = tid & 1;  // 2 threads/row, 32B each
      const u32x4v* sa = (const u32x4v*)((const char*)(A) +
          ((size_t)(m0 + row) * Kdim + kt * BK) * 2 + seg * 32);
      u32x4v* da = (u32x4v*)(As + row * 64 + seg * 32);
      da[0] = sa[0]; da[1] = sa[1];
      const u32x4v* sb = (const u32x4v*)((const char*)(Bt) +
          ((size_t)(n0 + row) * Kdim + kt * BK) * 2 + seg * 32);
      u32x4v* db = (u32x4v*)(Bs + row * 64 + seg * 32);
      db[0] = sb[0]; db[1] = sb[1];
    }
    __syncthreads();
    FragBf a[2], b[4];
#pragma unroll
    for (int i = 0; i < 2; ++i) {
      const char* p = As + (wm * 32 + i * 16 + lr) * 64 + half * 16;
      a[i].q[0] = *(const u32x4v*)p;
      a[i].q[1] = *(const u32x4v*)(p + 32);
    }
#pragma unroll
    for (int j = 0; j < 4; ++j) {
      const char* p = Bs + (wn * 64 + j * 16 + lr) * 64 + half * 16;
      b[j].q[0] = *(const u32x4v*)p;
      b[j].q[1] = *(const u32x4v*)(p + 32);
    }
#pragma unroll
    for (int i = 0; i < 2; ++i)
#pragma unroll
      for (int j = 0; j < 4; ++j)
        acc[i][j] = __builtin_amdgcn_wmma_f32_16x16x32_bf16(
            false, a[i].v, false, b[j].v, (short)0, acc[i][j], false, false);
    __syncthreads();
  }
#endif

  // Epilogue. C/D layout: VGPR r -> M=r+8*half, N=lr (per 16x16 f32 table).
#pragma unroll
  for (int i = 0; i < 2; ++i) {
#pragma unroll
    for (int j = 0; j < 4; ++j) {
#pragma unroll
      for (int r = 0; r < 8; ++r) {
        const int gr = m0 + wm * 32 + i * 16 + r + half * 8;
        const int gc = n0 + wn * 64 + j * 16 + lr;
        const size_t idx = (size_t)gr * Ncols + gc;
        const float s = acc[i][j][r];
        if (MODE == 0) {
          outBf[idx] = (__bf16)s;
        } else if (MODE == 1) {
          outF[idx] = s;
        } else {
          float uu = u[idx];
          const float a  = fmaxf(uu - SHRINK, 0.f);
          const float du = -uu + excite[idx] - s + a;  // LCA dynamics
          const float g  = -du;                        // u.grad
          const float mm = BETA1 * m[idx] + (1.f - BETA1) * g;
          const float vv = BETA2 * v[idx] + (1.f - BETA2) * g * g;
          uu = uu * (1.f - LRC * WDC);                 // decoupled weight decay
          uu -= LRC * (mm * c1) / (sqrtf(vv * c2) + EPSC);
          u[idx] = uu; m[idx] = mm; v[idx] = vv;
          actOut[idx] = (__bf16)fmaxf(uu - SHRINK, 0.f);
        }
      }
    }
  }
}

__global__ void cast_f32_bf16(const float* __restrict__ src,
                              __bf16* __restrict__ dst, int n) {
  int i = blockIdx.x * blockDim.x + threadIdx.x;
  if (i < n) dst[i] = (__bf16)src[i];
}

// Ft[n,d] = (bf16) F[d,n]   (F: D x K row-major) -- 32x32 LDS tile transpose
__global__ void transpose_cast(const float* __restrict__ F,
                               __bf16* __restrict__ Ft, int D, int K) {
  __shared__ float t[32][33];
  const int d0 = blockIdx.x * 32, n0 = blockIdx.y * 32;
  const int tx = threadIdx.x, ty = threadIdx.y;
#pragma unroll
  for (int i = 0; i < 32; i += 8)
    t[ty + i][tx] = F[(size_t)(d0 + ty + i) * K + (n0 + tx)];
  __syncthreads();
#pragma unroll
  for (int i = 0; i < 32; i += 8)
    Ft[(size_t)(n0 + ty + i) * D + (d0 + tx)] = (__bf16)t[tx][ty + i];
}

__global__ void init_state(float* u, float* m, float* v, __bf16* a0,
                           __bf16* a1, int n) {
  int i = blockIdx.x * blockDim.x + threadIdx.x;
  if (i < n) {
    u[i] = 0.f; m[i] = 0.f; v[i] = 0.f;
    a0[i] = (__bf16)0.f; a1[i] = (__bf16)0.f;
  }
}

__global__ void finalize(const float* __restrict__ u, float* __restrict__ out,
                         int n) {
  int i = blockIdx.x * blockDim.x + threadIdx.x;
  if (i < n) out[i] = fmaxf(u[i] - SHRINK, 0.f);
}

extern "C" void kernel_launch(void* const* d_in, const int* in_sizes, int n_in,
                              void* d_out, int out_size, void* d_ws,
                              size_t ws_size, hipStream_t stream) {
  (void)in_sizes; (void)n_in; (void)out_size; (void)ws_size;
  const int B = 512, D = 4096, K = 2048;
  const float* images  = (const float*)d_in[0];  // [B, D]
  const float* filters = (const float*)d_in[1];  // [D, K]
  float* out = (float*)d_out;                    // [B, K]

  char* ws = (char*)d_ws;
  size_t off = 0;
  auto carve = [&](size_t bytes) -> void* {
    void* p = ws + off;
    off += (bytes + 255) & ~(size_t)255;
    return p;
  };
  __bf16* Xb  = (__bf16*)carve((size_t)B * D * 2);   //  4 MB  X in bf16
  __bf16* Ft  = (__bf16*)carve((size_t)K * D * 2);   // 16 MB  F^T bf16
  __bf16* inh = (__bf16*)carve((size_t)K * K * 2);   //  8 MB  inhibit bf16
  float*  exc = (float*) carve((size_t)B * K * 4);   //  4 MB
  float*  u   = (float*) carve((size_t)B * K * 4);
  float*  mb  = (float*) carve((size_t)B * K * 4);
  float*  vb  = (float*) carve((size_t)B * K * 4);
  __bf16* act0 = (__bf16*)carve((size_t)B * K * 2);  // ping-pong act buffers
  __bf16* act1 = (__bf16*)carve((size_t)B * K * 2);
  __bf16* act[2] = {act0, act1};

  const int nBK = B * K;  // 1,048,576
  cast_f32_bf16<<<dim3((B * D + 255) / 256), dim3(256), 0, stream>>>(images, Xb, B * D);
  transpose_cast<<<dim3(D / 32, K / 32), dim3(32, 8), 0, stream>>>(filters, Ft, D, K);
  init_state<<<dim3((nBK + 255) / 256), dim3(256), 0, stream>>>(u, mb, vb, act0, act1, nBK);

  // inhibit = Ft @ Ft^T  (== F^T F),  [K x K]
  wmma_gemm<0><<<dim3(K / BN, K / BM), dim3(256), 0, stream>>>(
      Ft, Ft, D, K, K, K, inh, nullptr, nullptr, nullptr, nullptr, nullptr,
      nullptr, 0.f, 0.f);
  // excite = X @ F,  [B x K]
  wmma_gemm<1><<<dim3(K / BN, B / BM), dim3(256), 0, stream>>>(
      Xb, Ft, D, K, B, K, nullptr, exc, nullptr, nullptr, nullptr, nullptr,
      nullptr, 0.f, 0.f);

  float b1t = 1.f, b2t = 1.f;
  for (int t = 1; t <= NITER; ++t) {
    b1t *= BETA1; b2t *= BETA2;
    const float c1 = 1.f / (1.f - b1t);
    const float c2 = 1.f / (1.f - b2t);
    wmma_gemm<2><<<dim3(K / BN, B / BM), dim3(256), 0, stream>>>(
        act[(t + 1) & 1], inh, K, K, B, K, nullptr, nullptr, u, mb, vb, exc,
        act[t & 1], c1, c2);
  }

  finalize<<<dim3((nBK + 255) / 256), dim3(256), 0, stream>>>(u, out, nBK);
}
